// BEVFormerEncoder_58317065945168
// MI455X (gfx1250) — matrix operand order repaired and checked by reading, not compiled, not run on hardware
//
#include <hip/hip_runtime.h>

// ============================================================================
// BEVFormer encoder forward for gfx1250 (MI455X), bs = 1.
//
// Dense projections: bf16 WMMA (v_wmma_f32_16x16x32_bf16), fp32 accumulate.
// Weights are converted+transposed to bf16 [N][K] once per call, so GEMM
// staging is pure vector copies (no per-tile transpose scatter). Tile:
// BM=64 x BN=64 x BK=64, 8 waves, wave tile 16x32 -> 4 WMMAs per K-step.
// The workload is dominated by L2-resident bilinear gathers (~5 GB/layer at
// 23.3 TB/s-class bandwidth); GEMM FLOPs (~40 GF) are negligible, so bf16
// matrix path + fused epilogues is the right operating point.
// ============================================================================

typedef __bf16 bf16_t;
typedef bf16_t v16bf __attribute__((ext_vector_type(16)));
typedef float  v8f   __attribute__((ext_vector_type(8)));

#define NQ_      6400
#define C_       256
#define NH_      8
#define DH_      32
#define NCAM_    6
#define NV_      9520
#define DP_      4
#define FFN_     512
#define NLAYERS_ 3

__device__ __forceinline__ unsigned short f2bf(float f) {
  unsigned int u = __builtin_bit_cast(unsigned int, f);
  unsigned int r = u + 0x7FFFu + ((u >> 16) & 1u);
  return (unsigned short)(r >> 16);
}

union Frag  { v16bf v; uint4 q[2]; };
union Pack8 { unsigned short h[8]; uint4 q; };

// ---------------------------------------------------------------------------
// Weight convert + transpose: W (f32, [L][K][N]) -> Bt (bf16, [L][N][K])
// ---------------------------------------------------------------------------
__global__ void k_wtcvt(const float* __restrict__ W, bf16_t* __restrict__ Bt,
                        int K, int N, int total)
{
  int t = blockIdx.x * 256 + threadIdx.x;
  if (t >= total) return;
  int li = t / (K * N);
  int r  = t - li * (K * N);
  int k = r / N, n = r - k * N;
  unsigned short b = f2bf(W[t]);
  Bt[(size_t)li * N * K + (size_t)n * K + k] = __builtin_bit_cast(bf16_t, b);
}

// ---------------------------------------------------------------------------
// GEMM: C[M,N] = act(A[M,K] * B[K,N] + bias[N]) (+ resid)
// A fp32 row-major; Bt bf16 pre-transposed [N][K].
// Block 256 thr = 8 waves; tile 64x64, K-step 64; wave = 16x32 (4 WMMA/step).
// N % 64 == 0, K % 64 == 0 (always true here); M tail handled.
// ---------------------------------------------------------------------------
template <int RELU, int RES>
__global__ __launch_bounds__(256) void k_gemm_bf16(
    const float* __restrict__ A, const bf16_t* __restrict__ Bt,
    const float* __restrict__ bias, const float* __restrict__ resid,
    float* __restrict__ Cout, int M, int N, int K)
{
  __shared__ unsigned short sA[64 * 72];   // [row][k], stride 72 (144B, 16B-mult)
  __shared__ unsigned short sB[64 * 72];   // [n][k]

  const int m0   = blockIdx.y * 64;
  const int n0   = blockIdx.x * 64;
  const int tid  = threadIdx.x;
  const int wave = tid >> 5;
  const int lane = tid & 31;
  const int wm   = wave >> 1;        // 0..3 : 16-row group
  const int wn   = wave & 1;         // 0..1 : 32-col group
  const int half = lane >> 4;
  const int idx  = lane & 15;

  // staging coordinates: each thread owns 16 consecutive k of one row
  const int srow = tid >> 2;               // 0..63
  const int kgrp = (tid & 3) << 4;         // 0,16,32,48
  const int arow = (m0 + srow < M) ? (m0 + srow) : (M - 1);   // clamp tail

  v8f acc[2] = {};

  for (int k0 = 0; k0 < K; k0 += 64) {
    // ---- stage A (f32 -> bf16) : 4x float4 load, 2x uint4 LDS store ----
    {
      const float* ap = A + (size_t)arow * K + k0 + kgrp;
      float4 f0 = *(const float4*)(ap + 0);
      float4 f1 = *(const float4*)(ap + 4);
      float4 f2 = *(const float4*)(ap + 8);
      float4 f3 = *(const float4*)(ap + 12);
      Pack8 p0, p1;
      p0.h[0] = f2bf(f0.x); p0.h[1] = f2bf(f0.y); p0.h[2] = f2bf(f0.z); p0.h[3] = f2bf(f0.w);
      p0.h[4] = f2bf(f1.x); p0.h[5] = f2bf(f1.y); p0.h[6] = f2bf(f1.z); p0.h[7] = f2bf(f1.w);
      p1.h[0] = f2bf(f2.x); p1.h[1] = f2bf(f2.y); p1.h[2] = f2bf(f2.z); p1.h[3] = f2bf(f2.w);
      p1.h[4] = f2bf(f3.x); p1.h[5] = f2bf(f3.y); p1.h[6] = f2bf(f3.z); p1.h[7] = f2bf(f3.w);
      *(uint4*)&sA[srow * 72 + kgrp]     = p0.q;
      *(uint4*)&sA[srow * 72 + kgrp + 8] = p1.q;
    }
    // ---- stage B (already bf16, [n][k]) : pure uint4 copies ----
    {
      const bf16_t* bp = Bt + (size_t)(n0 + srow) * K + k0 + kgrp;
      uint4 b0 = *(const uint4*)(bp);
      uint4 b1 = *(const uint4*)(bp + 8);
      *(uint4*)&sB[srow * 72 + kgrp]     = b0;
      *(uint4*)&sB[srow * 72 + kgrp + 8] = b1;
    }
    __syncthreads();

    // ---- fragments ----
    Frag fa[2], fb[2][2];
    const unsigned short* rA = &sA[(wm * 16 + idx) * 72];
    // A lane layout: contiguous K = 16h..16h+15 per 32-K half
    fa[0].q[0] = *(const uint4*)(rA + 16 * half);
    fa[0].q[1] = *(const uint4*)(rA + 16 * half + 8);
    fa[1].q[0] = *(const uint4*)(rA + 32 + 16 * half);
    fa[1].q[1] = *(const uint4*)(rA + 32 + 16 * half + 8);
#pragma unroll
    for (int sn = 0; sn < 2; ++sn) {
      const unsigned short* rB = &sB[(wn * 32 + sn * 16 + idx) * 72];
      // B lane layout: contiguous K = 16h..16h+15 per 32-K half
      fb[sn][0].q[0] = *(const uint4*)(rB + 16 * half);
      fb[sn][0].q[1] = *(const uint4*)(rB + 16 * half + 8);
      fb[sn][1].q[0] = *(const uint4*)(rB + 32 + 16 * half);
      fb[sn][1].q[1] = *(const uint4*)(rB + 32 + 16 * half + 8);
    }
#pragma unroll
    for (int kh = 0; kh < 2; ++kh)
#pragma unroll
      for (int sn = 0; sn < 2; ++sn)
        acc[sn] = __builtin_amdgcn_wmma_f32_16x16x32_bf16(
            false, fa[kh].v, false, fb[sn][kh].v, (short)0, acc[sn], false, false);
    __syncthreads();
  }

  // ---- epilogue: D layout VGPR r -> M = r + 8*half, N = idx ----
  const int rowb = m0 + wm * 16 + half * 8;
  const bool full = (m0 + 64 <= M);
#pragma unroll
  for (int sn = 0; sn < 2; ++sn) {
    const int col = n0 + wn * 32 + sn * 16 + idx;
    const float bv = bias[col];
    size_t base = (size_t)rowb * N + col;
    if (full) {
#pragma unroll
      for (int r = 0; r < 8; ++r) {
        float v = acc[sn][r] + bv;
        if (RELU) v = fmaxf(v, 0.f);
        if (RES)  v += resid[base];
        Cout[base] = v;
        base += N;
      }
    } else {
#pragma unroll
      for (int r = 0; r < 8; ++r) {
        if (rowb + r < M) {
          float v = acc[sn][r] + bv;
          if (RELU) v = fmaxf(v, 0.f);
          if (RES)  v += resid[base];
          Cout[base] = v;
        }
        base += N;
      }
    }
  }
}

// ---------------------------------------------------------------------------
// lidar->image matrices: l2i = viewpad(K) @ inverse_rigid(ext) @ lid  (6 cams)
// ---------------------------------------------------------------------------
__global__ void k_l2i(const float* __restrict__ Kin, const float* __restrict__ ext,
                      const float* __restrict__ lid, float* __restrict__ l2i)
{
  int cam = threadIdx.x;
  if (cam >= NCAM_) return;
  const float* kk = Kin + cam * 9;
  float V[16] = {};
  V[0] = kk[0]; V[1] = kk[1]; V[2]  = kk[2];
  V[4] = kk[3]; V[5] = kk[4]; V[6]  = kk[5];
  V[8] = kk[6]; V[9] = kk[7]; V[10] = kk[8];
  V[15] = 1.f;
  const float* E = ext + cam * 16;
  float Inv[16] = {};
  for (int i = 0; i < 3; ++i)
    for (int j = 0; j < 3; ++j) Inv[i * 4 + j] = E[j * 4 + i];   // R^T
  float t0 = E[3], t1 = E[7], t2 = E[11];
  for (int i = 0; i < 3; ++i)
    Inv[i * 4 + 3] = -(E[i] * t0 + E[4 + i] * t1 + E[8 + i] * t2);
  Inv[15] = 1.f;
  float T1[16], Mo[16];
  for (int i = 0; i < 4; ++i)
    for (int j = 0; j < 4; ++j) {
      float s = 0.f;
      for (int k = 0; k < 4; ++k) s += Inv[i * 4 + k] * lid[k * 4 + j];
      T1[i * 4 + j] = s;
    }
  for (int i = 0; i < 4; ++i)
    for (int j = 0; j < 4; ++j) {
      float s = 0.f;
      for (int k = 0; k < 4; ++k) s += V[i * 4 + k] * T1[k * 4 + j];
      Mo[i * 4 + j] = s;
    }
  for (int i = 0; i < 16; ++i) l2i[cam * 16 + i] = Mo[i];
}

// ---------------------------------------------------------------------------
// Per (cam, q): project 4 pillar points -> normalized image coords + hit mask
// ---------------------------------------------------------------------------
__global__ void k_refcam(const float* __restrict__ l2i, float* __restrict__ refcam,
                         float* __restrict__ hit)
{
  int t = blockIdx.x * 256 + threadIdx.x;
  if (t >= NCAM_ * NQ_) return;
  int cam = t / NQ_, q = t % NQ_;
  int iy = q / 80, ix = q % 80;
  const float* M = l2i + cam * 16;
  float wx = ((ix + 0.5f) * (1.f / 80.f)) * 102.4f - 51.2f;
  float wy = ((iy + 0.5f) * (1.f / 80.f)) * 102.4f - 51.2f;
  float h = 0.f;
  for (int d = 0; d < DP_; ++d) {
    float z01 = (0.5f + (float)d * (7.f / 3.f)) * 0.125f;   // linspace(.5,7.5,4)/8
    float wz = z01 * 8.f - 5.f;
    float px = M[0] * wx + M[1] * wy + M[2]  * wz + M[3];
    float py = M[4] * wx + M[5] * wy + M[6]  * wz + M[7];
    float pz = M[8] * wx + M[9] * wy + M[10] * wz + M[11];
    int m = pz > 1e-5f;
    float zc = fmaxf(pz, 1e-5f);
    float sxp = px / zc * (1.f / 896.f);
    float syp = py / zc * (1.f / 512.f);
    m = m && (sxp > 0.f) && (sxp < 1.f) && (syp > 0.f) && (syp < 1.f);
    refcam[(size_t)t * 8 + d * 2]     = sxp;
    refcam[(size_t)t * 8 + d * 2 + 1] = syp;
    if (m) h = 1.f;
  }
  hit[t] = h;
}

// qcat[q] = concat(prev_bev[q], x[q] + pos[q])
__global__ void k_build_qcat(const float* __restrict__ prev, const float* __restrict__ x,
                             const float* __restrict__ pos, float* __restrict__ qcat, int n)
{
  int t = blockIdx.x * 256 + threadIdx.x;
  if (t >= n) return;
  int q = t >> 8, c = t & 255;
  qcat[(size_t)q * 512 + c]       = prev[t];
  qcat[(size_t)q * 512 + 256 + c] = x[t] + pos[t];
}

__global__ void k_softmax4(float* __restrict__ aw, int n) {
  int t = blockIdx.x * 256 + threadIdx.x;
  if (t >= n) return;
  float* p = aw + (size_t)t * 4;
  float m = fmaxf(fmaxf(p[0], p[1]), fmaxf(p[2], p[3]));
  float e0 = expf(p[0] - m), e1 = expf(p[1] - m), e2 = expf(p[2] - m), e3 = expf(p[3] - m);
  float s = 1.f / (e0 + e1 + e2 + e3);
  p[0] = e0 * s; p[1] = e1 * s; p[2] = e2 * s; p[3] = e3 * s;
}

__global__ void k_softmax32(float* __restrict__ aw, int n) {
  int t = blockIdx.x * 256 + threadIdx.x;
  if (t >= n) return;
  float* p = aw + (size_t)t * 32;
  float m = -1e30f;
  for (int i = 0; i < 32; ++i) m = fmaxf(m, p[i]);
  float s = 0.f;
  float e[32];
  for (int i = 0; i < 32; ++i) { e[i] = expf(p[i] - m); s += e[i]; }
  s = 1.f / s;
  for (int i = 0; i < 32; ++i) p[i] = e[i] * s;
}

// bilinear sample with zero padding outside; lane carries channel -> coalesced
__device__ __forceinline__ float bilin(const float* __restrict__ v, float x, float y,
                                       int h, int w, int coff)
{
  float xf = floorf(x), yf = floorf(y);
  int x0 = (int)xf, y0 = (int)yf;
  float lx = x - xf, ly = y - yf;
  float r = 0.f;
#pragma unroll
  for (int dy = 0; dy < 2; ++dy)
#pragma unroll
    for (int dx = 0; dx < 2; ++dx) {
      int xi = x0 + dx, yi = y0 + dy;
      float wt = (dx ? lx : 1.f - lx) * (dy ? ly : 1.f - ly);
      if (xi >= 0 && xi < w && yi >= 0 && yi < h)
        r += wt * v[(size_t)(yi * w + xi) * C_ + coff];
    }
  return r;
}

// Temporal self-attention deformable sampling: wave per (q, head), lane=channel
__global__ __launch_bounds__(256) void k_msda_tsa(
    const float* __restrict__ v2, const float* __restrict__ off,
    const float* __restrict__ aw, const float* __restrict__ shift,
    float* __restrict__ out)
{
  int wid  = blockIdx.x * 8 + (threadIdx.x >> 5);
  int lane = threadIdx.x & 31;
  int q = wid >> 3, head = wid & 7;
  int iy = q / 80, ix = q % 80;
  float sx = shift[0], sy = shift[1];
  float acc = 0.f;
  for (int b = 0; b < 2; ++b) {
    float rx = (ix + 0.5f) * (1.f / 80.f) + (b == 0 ? sx : 0.f);
    float ry = (iy + 0.5f) * (1.f / 80.f) + (b == 0 ? sy : 0.f);
    const float* vb = v2 + (size_t)b * NQ_ * C_;
    for (int p = 0; p < 4; ++p) {
      float ox = off[(size_t)q * 128 + head * 16 + b * 8 + p * 2]     * (1.f / 80.f);
      float oy = off[(size_t)q * 128 + head * 16 + b * 8 + p * 2 + 1] * (1.f / 80.f);
      float w  = aw[(size_t)q * 64 + head * 8 + b * 4 + p];
      float x = (rx + ox) * 80.f - 0.5f;
      float y = (ry + oy) * 80.f - 0.5f;
      acc += 0.5f * w * bilin(vb, x, y, 80, 80, head * 32 + lane);
    }
  }
  out[(size_t)q * C_ + head * 32 + lane] = acc;
}

// Spatial cross-attention: wave per (q, head); loops cams/levels/points;
// hit-masking and divide-by-count fused (no atomics).
__global__ __launch_bounds__(256) void k_msda_sca(
    const float* __restrict__ vv, const float* __restrict__ off,
    const float* __restrict__ aw, const float* __restrict__ refcam,
    const float* __restrict__ hit, float* __restrict__ out)
{
  const int lvlH[4] = {64, 32, 16, 8};
  const int lvlW[4] = {112, 56, 28, 14};
  const int lvlS[4] = {0, 7168, 8960, 9408};
  int wid  = blockIdx.x * 8 + (threadIdx.x >> 5);
  int lane = threadIdx.x & 31;
  int q = wid >> 3, head = wid & 7;
  float acc = 0.f, cnt = 0.f;
  for (int cam = 0; cam < NCAM_; ++cam) {
    float hv = hit[cam * NQ_ + q];
    cnt += hv;
    if (hv == 0.f) continue;                    // uniform across the wave
    const float* vcam = vv + (size_t)cam * NV_ * C_;
    const float* rc = refcam + ((size_t)cam * NQ_ + q) * 8;
    for (int l = 0; l < 4; ++l) {
      float ws = (float)lvlW[l], hs = (float)lvlH[l];
      const float* vlev = vcam + (size_t)lvlS[l] * C_;
      for (int p = 0; p < 8; ++p) {
        int d = p & 3;                          // p = rep*4 + d (pillar index)
        float rx = rc[d * 2], ry = rc[d * 2 + 1];
        float ox = off[(size_t)q * 512 + head * 64 + l * 16 + p * 2]     / ws;
        float oy = off[(size_t)q * 512 + head * 64 + l * 16 + p * 2 + 1] / hs;
        float w  = aw[(size_t)q * 256 + head * 32 + l * 8 + p];
        float x = (rx + ox) * ws - 0.5f;
        float y = (ry + oy) * hs - 0.5f;
        acc += w * bilin(vlev, x, y, lvlH[l], lvlW[l], head * 32 + lane);
      }
    }
  }
  cnt = fmaxf(cnt, 1.f);
  out[(size_t)q * C_ + head * 32 + lane] = acc / cnt;
}

// LayerNorm over C=256, wave per row (wave32), shuffle reductions
__global__ __launch_bounds__(256) void k_layernorm(
    float* __restrict__ x, const float* __restrict__ g, const float* __restrict__ b)
{
  int row  = blockIdx.x * 8 + (threadIdx.x >> 5);
  int lane = threadIdx.x & 31;
  float* px = x + (size_t)row * C_;
  float v[8];
  float s = 0.f;
#pragma unroll
  for (int j = 0; j < 8; ++j) { v[j] = px[lane + 32 * j]; s += v[j]; }
#pragma unroll
  for (int o = 16; o > 0; o >>= 1) s += __shfl_xor(s, o, 32);
  float mean = s * (1.f / 256.f);
  float qv = 0.f;
#pragma unroll
  for (int j = 0; j < 8; ++j) { float d = v[j] - mean; qv += d * d; }
#pragma unroll
  for (int o = 16; o > 0; o >>= 1) qv += __shfl_xor(qv, o, 32);
  float rstd = rsqrtf(qv * (1.f / 256.f) + 1e-5f);
#pragma unroll
  for (int j = 0; j < 8; ++j) {
    int c = lane + 32 * j;
    px[c] = (v[j] - mean) * rstd * g[c] + b[c];
  }
}

// ===========================================================================
extern "C" void kernel_launch(void* const* d_in, const int* in_sizes, int n_in,
                              void* d_out, int out_size, void* d_ws, size_t ws_size,
                              hipStream_t stream)
{
  // --- inputs (setup_inputs order; 'params' pytree flattened in sorted-key
  //     leaf order: ffn{W1,W2,b1,b2}, ln{b0,b1,b2,g0,g1,g2},
  //     sca{Wattn,Woff,Wout,Wv,battn,boff,bout,bv}, tsa{same}) ---
  const float* bev_query = (const float*)d_in[0];
  const float* value     = (const float*)d_in[1];
  const float* bev_pos   = (const float*)d_in[4];
  const float* prev_bev  = (const float*)d_in[7];
  const float* shift     = (const float*)d_in[8];
  const float* camK      = (const float*)d_in[11];
  const float* camE      = (const float*)d_in[12];
  const float* lidE      = (const float*)d_in[13];
  const float* fW1  = (const float*)d_in[14];
  const float* fW2  = (const float*)d_in[15];
  const float* fb1  = (const float*)d_in[16];
  const float* fb2  = (const float*)d_in[17];
  const float* lnb0 = (const float*)d_in[18];
  const float* lnb1 = (const float*)d_in[19];
  const float* lnb2 = (const float*)d_in[20];
  const float* lng0 = (const float*)d_in[21];
  const float* lng1 = (const float*)d_in[22];
  const float* lng2 = (const float*)d_in[23];
  const float* sWattn = (const float*)d_in[24];
  const float* sWoff  = (const float*)d_in[25];
  const float* sWout  = (const float*)d_in[26];
  const float* sWv    = (const float*)d_in[27];
  const float* sbattn = (const float*)d_in[28];
  const float* sboff  = (const float*)d_in[29];
  const float* sbout  = (const float*)d_in[30];
  const float* sbv    = (const float*)d_in[31];
  const float* tWattn = (const float*)d_in[32];
  const float* tWoff  = (const float*)d_in[33];
  const float* tWout  = (const float*)d_in[34];
  const float* tWv    = (const float*)d_in[35];
  const float* tbattn = (const float*)d_in[36];
  const float* tboff  = (const float*)d_in[37];
  const float* tbout  = (const float*)d_in[38];
  const float* tbv    = (const float*)d_in[39];

  // --- workspace carve-up: fp32 buffers (reused across phases) + bf16 arena ---
  float* ws = (float*)d_ws;
  size_t o = 0;
  auto carve = [&](size_t n) { float* p = ws + o; o += (n + 63) & ~(size_t)63; return p; };
  float* xbuf    = carve((size_t)NQ_ * C_);          // running BEV state
  float* vbig    = carve((size_t)NCAM_ * NV_ * C_);  // TSA: v2 (2*NQ*C); SCA: vv
  float* qbuf    = carve((size_t)NQ_ * 512);         // qcat / sca_off / ffn hidden
  float* offaw   = carve((size_t)NQ_ * 256);         // TSA: off(128)+aw(64); SCA: aw(256)
  float* attnout = carve((size_t)NQ_ * C_);          // MSDA output
  float* refc    = carve((size_t)NCAM_ * NQ_ * 8);
  float* hitb    = carve((size_t)NCAM_ * NQ_);
  float* l2ib    = carve((size_t)NCAM_ * 16);

  bf16_t* warena = (bf16_t*)(ws + o);
  size_t ob = 0;
  auto carveb = [&](size_t n) { bf16_t* p = warena + ob; ob += (n + 63) & ~(size_t)63; return p; };
  bf16_t* tWv_b   = carveb((size_t)NLAYERS_ * 256 * 256);
  bf16_t* tWoff_b = carveb((size_t)NLAYERS_ * 512 * 128);
  bf16_t* tWat_b  = carveb((size_t)NLAYERS_ * 512 * 64);
  bf16_t* tWo_b   = carveb((size_t)NLAYERS_ * 256 * 256);
  bf16_t* sWv_b   = carveb((size_t)NLAYERS_ * 256 * 256);
  bf16_t* sWoff_b = carveb((size_t)NLAYERS_ * 256 * 512);
  bf16_t* sWat_b  = carveb((size_t)NLAYERS_ * 256 * 256);
  bf16_t* sWo_b   = carveb((size_t)NLAYERS_ * 256 * 256);
  bf16_t* fW1_b   = carveb((size_t)NLAYERS_ * 256 * 512);
  bf16_t* fW2_b   = carveb((size_t)NLAYERS_ * 512 * 256);
  (void)ws_size; (void)in_sizes; (void)n_in; (void)out_size;

  float* offT = offaw;                      // TSA offsets (NQ x 128)
  float* awT  = offaw + (size_t)NQ_ * 128;  // TSA attn weights (NQ x 64)

  auto gemm = [&](const float* A, const bf16_t* Bt, const float* bias,
                  const float* resid, float* Cp, int M, int N, int K, int relu) {
    dim3 g((unsigned)(N / 64), (unsigned)((M + 63) / 64));
    if (resid)
      k_gemm_bf16<0, 1><<<g, 256, 0, stream>>>(A, Bt, bias, resid, Cp, M, N, K);
    else if (relu)
      k_gemm_bf16<1, 0><<<g, 256, 0, stream>>>(A, Bt, bias, nullptr, Cp, M, N, K);
    else
      k_gemm_bf16<0, 0><<<g, 256, 0, stream>>>(A, Bt, bias, nullptr, Cp, M, N, K);
  };
  auto wtcvt = [&](const float* W, bf16_t* Bt, int K, int N) {
    int total = NLAYERS_ * K * N;
    k_wtcvt<<<(total + 255) / 256, 256, 0, stream>>>(W, Bt, K, N, total);
  };

  // --- one-time per call: weight convert/transpose + geometry ---
  wtcvt(tWv,   tWv_b,   256, 256);
  wtcvt(tWoff, tWoff_b, 512, 128);
  wtcvt(tWattn,tWat_b,  512,  64);
  wtcvt(tWout, tWo_b,   256, 256);
  wtcvt(sWv,   sWv_b,   256, 256);
  wtcvt(sWoff, sWoff_b, 256, 512);
  wtcvt(sWattn,sWat_b,  256, 256);
  wtcvt(sWout, sWo_b,   256, 256);
  wtcvt(fW1,   fW1_b,   256, 512);
  wtcvt(fW2,   fW2_b,   512, 256);

  hipMemcpyAsync(xbuf, bev_query, (size_t)NQ_ * C_ * sizeof(float),
                 hipMemcpyDeviceToDevice, stream);
  k_l2i<<<1, 32, 0, stream>>>(camK, camE, lidE, l2ib);
  k_refcam<<<(NCAM_ * NQ_ + 255) / 256, 256, 0, stream>>>(l2ib, refc, hitb);

  for (int li = 0; li < NLAYERS_; ++li) {
    // ---------------- Temporal self-attention ----------------
    gemm(prev_bev,  tWv_b + (size_t)li * 65536, tbv + li * 256, nullptr,
         vbig,                    NQ_, 256, 256, 0);
    gemm(bev_query, tWv_b + (size_t)li * 65536, tbv + li * 256, nullptr,
         vbig + (size_t)NQ_ * C_, NQ_, 256, 256, 0);
    k_build_qcat<<<NQ_, 256, 0, stream>>>(prev_bev, xbuf, bev_pos, qbuf, NQ_ * C_);
    gemm(qbuf, tWoff_b + (size_t)li * 65536, tboff + li * 128, nullptr, offT, NQ_, 128, 512, 0);
    gemm(qbuf, tWat_b  + (size_t)li * 32768, tbattn + li * 64, nullptr, awT,  NQ_,  64, 512, 0);
    k_softmax4<<<(NQ_ * 16 + 255) / 256, 256, 0, stream>>>(awT, NQ_ * 16);
    k_msda_tsa<<<NQ_, 256, 0, stream>>>(vbig, offT, awT, shift, attnout);
    gemm(attnout, tWo_b + (size_t)li * 65536, tbout + li * 256, xbuf, xbuf, NQ_, 256, 256, 0);
    k_layernorm<<<NQ_ / 8, 256, 0, stream>>>(xbuf, lng0 + li * 256, lnb0 + li * 256);

    // ---------------- Spatial cross-attention ----------------
    gemm(value, sWv_b + (size_t)li * 65536, sbv + li * 256, nullptr,
         vbig, NCAM_ * NV_, 256, 256, 0);
    // offsets / attention weights are camera-independent: compute once
    gemm(xbuf, sWoff_b + (size_t)li * 131072, sboff + li * 512, nullptr, qbuf,  NQ_, 512, 256, 0);
    gemm(xbuf, sWat_b  + (size_t)li * 65536,  sbattn + li * 256, nullptr, offaw, NQ_, 256, 256, 0);
    k_softmax32<<<(NQ_ * NH_ + 255) / 256, 256, 0, stream>>>(offaw, NQ_ * NH_);
    k_msda_sca<<<NQ_, 256, 0, stream>>>(vbig, qbuf, offaw, refc, hitb, attnout);
    gemm(attnout, sWo_b + (size_t)li * 65536, sbout + li * 256, xbuf, xbuf, NQ_, 256, 256, 0);
    k_layernorm<<<NQ_ / 8, 256, 0, stream>>>(xbuf, lng1 + li * 256, lnb1 + li * 256);

    // ---------------- FFN ----------------
    gemm(xbuf, fW1_b + (size_t)li * 131072, fb1 + li * 512, nullptr, qbuf, NQ_, 512, 256, 1);
    gemm(qbuf, fW2_b + (size_t)li * 131072, fb2 + li * 256, xbuf,    xbuf, NQ_, 256, 512, 0);
    k_layernorm<<<NQ_ / 8, 256, 0, stream>>>(xbuf, lng2 + li * 256, lnb2 + li * 256);
  }

  hipMemcpyAsync(d_out, xbuf, (size_t)NQ_ * C_ * sizeof(float),
                 hipMemcpyDeviceToDevice, stream);
}
